// SCA_80187039416711
// MI455X (gfx1250) — matrix operand order repaired and checked by reading, not compiled, not run on hardware
//
#include <hip/hip_runtime.h>

// ---------------------------------------------------------------------------
// Types for CDNA5 WMMA (wave32)
// ---------------------------------------------------------------------------
typedef __attribute__((ext_vector_type(16))) __bf16 v16bf;
typedef __attribute__((ext_vector_type(8)))  __bf16 v8bf;
typedef __attribute__((ext_vector_type(8)))  float  v8f;

__device__ inline v8f wmma_bf16(v16bf a, v16bf b, v8f c) {
  // (neg_a, A, neg_b, B, c_mod, C, reuse_a, reuse_b)
  return __builtin_amdgcn_wmma_f32_16x16x32_bf16(false, a, false, b, (short)0, c,
                                                 false, false);
}

__device__ inline v16bf cat16(v8bf lo, v8bf hi) {
  v16bf r;
#pragma unroll
  for (int i = 0; i < 8; ++i) { r[i] = lo[i]; r[i + 8] = hi[i]; }
  return r;
}

__device__ inline v16bf ld16(const __bf16* p, int gap) {  // two v8bf, gap elems apart
  return cat16(*reinterpret_cast<const v8bf*>(p),
               *reinterpret_cast<const v8bf*>(p + gap));
}

#define HWD 4096
#define CCH 256
#define NB  4

// ---------------------------------------------------------------------------
// Elementwise converters / builders
// ---------------------------------------------------------------------------
__global__ void sca_cvt_bf16(const float* __restrict__ s, __bf16* __restrict__ d, int n) {
  int i = blockIdx.x * 256 + threadIdx.x;
  int stride = gridDim.x * 256;
  for (; i < n; i += stride) d[i] = (__bf16)s[i];
}

__global__ void sca_build_wcat(const float* __restrict__ wout,
                               const float* __restrict__ wsk,
                               __bf16* __restrict__ wcat) {
  int i = blockIdx.x * 256 + threadIdx.x;  // 256*512 elems
  if (i < 256 * 512) {
    int o = i >> 9, k = i & 511;
    float v = (k < 256) ? wout[o * 256 + k] : wsk[o * 256 + (k - 256)];
    wcat[i] = (__bf16)v;
  }
}

__global__ void sca_build_bsum(const float* __restrict__ bo,
                               const float* __restrict__ bs,
                               float* __restrict__ bsum) {
  int i = threadIdx.x;
  if (i < 256) bsum[i] = bo[i] + bs[i];
}

// ---------------------------------------------------------------------------
// LDS-tiled transpose + f32->bf16:  X [B][C][HW]  ->  XT [B][HW][C]
// grid (HW/64, C/32, B), block 256
// ---------------------------------------------------------------------------
__global__ __launch_bounds__(256) void sca_transpose_cvt(
    const float* __restrict__ X, __bf16* __restrict__ XT) {
  __shared__ __bf16 T[64 * 40];  // [hw][c], padded stride
  const int b = blockIdx.z;
  const int hw0 = blockIdx.x * 64, c0 = blockIdx.y * 32;
  const size_t ib = (size_t)b * CCH * HWD;
  const int tid = threadIdx.x;

#pragma unroll
  for (int it = 0; it < 8; ++it) {
    int id = tid + it * 256;          // 0..2047
    int hw = id & 63, c = id >> 6;    // coalesced along hw
    T[hw * 40 + c] = (__bf16)X[ib + (size_t)(c0 + c) * HWD + hw0 + hw];
  }
  __syncthreads();
  {
    int hw = tid >> 2, c8 = (tid & 3) * 8;  // one v8bf per thread
    v8bf v;
#pragma unroll
    for (int i = 0; i < 8; ++i) v[i] = T[hw * 40 + c8 + i];
    *reinterpret_cast<v8bf*>(XT + ib + (size_t)(hw0 + hw) * CCH + c0 + c8) = v;
  }
}

// ---------------------------------------------------------------------------
// conv1x1 GEMM, zero-LDS:  Y[b,o,n] = sum_k W[o,k] * XT[b,n,k] (+ bias)
// B input is n-major so every B-fragment is 16 contiguous bf16 per lane.
// Block = 256 threads = 8 waves, tile M=64 x N=128, K-step 32.
// MODE 0: f32 [o][n] + bias.  MODE 1: bf16 [n][o].  MODE 2: bf16 [o][n].
// ---------------------------------------------------------------------------
template <int MODE>
__global__ __launch_bounds__(256) void sca_gemm1x1(
    const __bf16* __restrict__ W, const __bf16* __restrict__ XT,
    const float* __restrict__ bias, float* __restrict__ Yf,
    __bf16* __restrict__ Yo, int M, int N, int K) {
  const int tid  = threadIdx.x;
  const int lane = tid & 31;
  const int wid  = tid >> 5;
  const int wrow = wid >> 1, wcol = wid & 1;
  const int hi   = lane >> 4;
  const int l16  = lane & 15;
  const int b    = blockIdx.z;
  const int bn0  = blockIdx.x * 128 + wcol * 64;
  const int m0   = blockIdx.y * 64 + wrow * 16;
  const size_t Xb = (size_t)b * N * K;

  v8f acc[4] = {};
  const __bf16* brow = XT + Xb + (size_t)l16 * K + hi * 16;  // + n*K + k0

  // software pipeline: preload fragments for k0 = 0
  v16bf a = ld16(W + (size_t)(m0 + l16) * K + hi * 16, 8);
  v16bf bb0 = ld16(brow + (size_t)(bn0 + 0 * 16) * K, 8);

  for (int k0 = 0; k0 < K; k0 += 32) {
    const int kn = k0 + 32;
    v16bf b1 = ld16(brow + (size_t)(bn0 + 1 * 16) * K + k0, 8);
    acc[0] = wmma_bf16(a, bb0, acc[0]);
    v16bf b2 = ld16(brow + (size_t)(bn0 + 2 * 16) * K + k0, 8);
    acc[1] = wmma_bf16(a, b1, acc[1]);
    v16bf b3 = ld16(brow + (size_t)(bn0 + 3 * 16) * K + k0, 8);
    acc[2] = wmma_bf16(a, b2, acc[2]);
    if (kn < K) {  // preload next step while last wmma runs
      bb0 = ld16(brow + (size_t)(bn0 + 0 * 16) * K + kn, 8);
      v16bf an = ld16(W + (size_t)(m0 + l16) * K + kn + hi * 16, 8);
      acc[3] = wmma_bf16(a, b3, acc[3]);
      a = an;
    } else {
      acc[3] = wmma_bf16(a, b3, acc[3]);
    }
  }

  // D layout: VGPR r -> M=r (lanes<16) / r+8 (lanes>=16), N=l16
  if constexpr (MODE == 0) {
    const size_t Yb = (size_t)b * M * N;
#pragma unroll
    for (int t = 0; t < 4; ++t) {
      int n = bn0 + t * 16 + l16;
#pragma unroll
      for (int r = 0; r < 8; ++r) {
        int m = m0 + r + 8 * hi;
        Yf[Yb + (size_t)m * N + n] = acc[t][r] + bias[m];
      }
    }
  } else if constexpr (MODE == 1) {
    const size_t Yb = (size_t)b * N * M;
#pragma unroll
    for (int t = 0; t < 4; ++t) {
      int n = bn0 + t * 16 + l16;
#pragma unroll
      for (int r = 0; r < 8; ++r) {
        int m = m0 + r + 8 * hi;
        Yo[Yb + (size_t)n * M + m] = (__bf16)acc[t][r];
      }
    }
  } else {
    const size_t Yb = (size_t)b * M * N;
#pragma unroll
    for (int t = 0; t < 4; ++t) {
      int n = bn0 + t * 16 + l16;
#pragma unroll
      for (int r = 0; r < 8; ++r) {
        int m = m0 + r + 8 * hi;
        Yo[Yb + (size_t)m * N + n] = (__bf16)acc[t][r];
      }
    }
  }
}

// ---------------------------------------------------------------------------
// GroupNorm over [b, group of 8 channels, 4096].  One block per (b,g).
// Emits n-major bf16 output Gt[b][hw][256]; optionally a raw bf16 copy into
// the concat buffer xcatT[b][hw][512] at columns 256+c0 (skip path).
// ---------------------------------------------------------------------------
__global__ __launch_bounds__(256) void sca_groupnorm(
    const float* __restrict__ Y, const float* __restrict__ gamma,
    const float* __restrict__ beta, __bf16* __restrict__ Gt,
    __bf16* __restrict__ Raw) {
  const int blk = blockIdx.x;
  const int b = blk >> 5, g = blk & 31;
  const int c0 = g * 8;
  const size_t base = ((size_t)b * CCH + c0) * HWD;
  const int tid = threadIdx.x;

  float s = 0.f, s2 = 0.f;
  for (int i = tid; i < 8 * HWD; i += 256) {
    float v = Y[base + i];
    s += v; s2 += v * v;
  }
#pragma unroll
  for (int off = 16; off; off >>= 1) {
    s  += __shfl_xor(s, off);
    s2 += __shfl_xor(s2, off);
  }
  __shared__ float sm[16];
  int wid = tid >> 5, lane = tid & 31;
  if (lane == 0) { sm[wid] = s; sm[8 + wid] = s2; }
  __syncthreads();
  if (tid == 0) {
    float ts = 0.f, t2 = 0.f;
#pragma unroll
    for (int i = 0; i < 8; ++i) { ts += sm[i]; t2 += sm[8 + i]; }
    float mu  = ts * (1.f / 32768.f);
    float var = t2 * (1.f / 32768.f) - mu * mu;
    sm[0] = mu;
    sm[1] = rsqrtf(var + 1e-5f);
  }
  __syncthreads();
  const float mu = sm[0], inv = sm[1];

  float gam[8], bet[8];
#pragma unroll
  for (int ci = 0; ci < 8; ++ci) { gam[ci] = gamma[c0 + ci] * inv; bet[ci] = beta[c0 + ci]; }

  const size_t gB = (size_t)b * HWD * CCH;
  const size_t rB = (size_t)b * HWD * 512;
  for (int hw = tid; hw < HWD; hw += 256) {
    v8bf gv, rv;
#pragma unroll
    for (int ci = 0; ci < 8; ++ci) {
      float v = Y[base + (size_t)ci * HWD + hw];  // coalesced across threads
      gv[ci] = (__bf16)((v - mu) * gam[ci] + bet[ci]);
      rv[ci] = (__bf16)v;
    }
    *reinterpret_cast<v8bf*>(Gt + gB + (size_t)hw * CCH + c0) = gv;
    if (Raw)
      *reinterpret_cast<v8bf*>(Raw + rB + (size_t)hw * 512 + 256 + c0) = rv;
  }
}

// ---------------------------------------------------------------------------
// Flash attention, zero block-wide LDS staging.
//   Qt, Kt : [B][HW][C] bf16 (n-major)  -> A / B frags are contiguous loads
//   Vb     : [B][C][HW] bf16 (natural)  -> PV B-frags are contiguous loads
//   OT     : xcatT [B][HW][512], attention fills columns 0..255
// Block = 128 threads (4 waves); wave owns 16 queries, full 256-c output tile.
// Per 32-key step: 16 WMMA (scores) + 16 WMMA (PV).
// Score loop is c-outer / h-inner so each load-pair is covered by two
// back-to-back WMMAs on independent accumulators; PV loop preloads V one
// tile ahead.
// ---------------------------------------------------------------------------
__global__ __launch_bounds__(128) void sca_flash_attn(
    const __bf16* __restrict__ Qt, const __bf16* __restrict__ Kt,
    const __bf16* __restrict__ Vb, __bf16* __restrict__ OT) {
  __shared__ __bf16 Psh[4][16 * 32];  // per-wave P tile [q][k]

  const int tid  = threadIdx.x;
  const int lane = tid & 31;
  const int wid  = tid >> 5;
  const int hi   = lane >> 4;
  const int l16  = lane & 15;
  const int b    = blockIdx.y;
  const int q0   = blockIdx.x * 64 + wid * 16;
  const size_t tB = (size_t)b * HWD * CCH;

  // preload Q A-fragments for the whole channel range: 8 x v16bf
  v16bf aQ[8];
  {
    const __bf16* qr = Qt + tB + (size_t)(q0 + l16) * CCH + hi * 16;
#pragma unroll
    for (int cc = 0; cc < 8; ++cc)
      aQ[cc] = ld16(qr + cc * 32, 8);
  }

  v8f acc[16] = {};
  float mrow[8], lrow[8];
#pragma unroll
  for (int r = 0; r < 8; ++r) { mrow[r] = -1e30f; lrow[r] = 0.f; }

  const __bf16* krow = Kt + tB + (size_t)l16 * CCH + hi * 16;  // + kpos*CCH + cc*32
  const __bf16* vrow = Vb + tB + (size_t)l16 * HWD + hi * 16;  // + c*HWD + k0

  for (int k0 = 0; k0 < HWD; k0 += 32) {
    // prefetch next key/value tiles into cache while we compute this step
    if (k0 + 32 < HWD) {
#pragma unroll
      for (int it = 0; it < 2; ++it) {
        int idx = tid + it * 128;  // 0..255
        __builtin_prefetch(Kt + tB + (size_t)(k0 + 32 + (idx >> 3)) * CCH +
                               (idx & 7) * 32, 0, 1);
        __builtin_prefetch(Vb + tB + (size_t)idx * HWD + k0 + 32, 0, 1);
      }
    }

    // scores: two 16q x 16k tiles; c-chunk outer, both tiles per chunk
    v8f S0 = {}, S1 = {};
    const __bf16* kr0 = krow + (size_t)k0 * CCH;
    const __bf16* kr1 = krow + (size_t)(k0 + 16) * CCH;
#pragma unroll
    for (int cc = 0; cc < 8; ++cc) {
      v16bf b0 = ld16(kr0 + cc * 32, 8);
      v16bf b1 = ld16(kr1 + cc * 32, 8);
      S0 = wmma_bf16(aQ[cc], b0, S0);
      S1 = wmma_bf16(aQ[cc], b1, S1);
    }

    // online softmax (rows live across the 16 lanes of each half)
    float alpha[8];
#pragma unroll
    for (int r = 0; r < 8; ++r) {
      float a0 = S0[r] * 0.0625f;  // 1/sqrt(256)
      float a1 = S1[r] * 0.0625f;
      float rm = fmaxf(a0, a1);
#pragma unroll
      for (int off = 8; off; off >>= 1) rm = fmaxf(rm, __shfl_xor(rm, off));
      float mn = fmaxf(mrow[r], rm);
      alpha[r] = __expf(mrow[r] - mn);
      mrow[r]  = mn;
      float p0 = __expf(a0 - mn);
      float p1 = __expf(a1 - mn);
      S0[r] = p0; S1[r] = p1;
      float rs = p0 + p1;
#pragma unroll
      for (int off = 8; off; off >>= 1) rs += __shfl_xor(rs, off);
      lrow[r] = lrow[r] * alpha[r] + rs;
    }
#pragma unroll
    for (int t = 0; t < 16; ++t)
#pragma unroll
      for (int r = 0; r < 8; ++r) acc[t][r] *= alpha[r];

    // P (C-layout) -> per-wave LDS [q][k] -> A-layout fragment
#pragma unroll
    for (int r = 0; r < 8; ++r) {
      Psh[wid][(r + 8 * hi) * 32 + l16]      = (__bf16)S0[r];
      Psh[wid][(r + 8 * hi) * 32 + 16 + l16] = (__bf16)S1[r];
    }
    asm volatile("s_wait_dscnt 0" ::: "memory");
    const __bf16* pr = &Psh[wid][l16 * 32 + hi * 16];
    v16bf ap = cat16(*(const v8bf*)pr, *(const v8bf*)(pr + 8));

    // PV: 16 c-tiles, V straight from global; pipeline V one tile ahead
    v16bf bv = ld16(vrow + (size_t)0 * 16 * HWD + k0, 8);
#pragma unroll
    for (int t = 0; t < 16; ++t) {
      v16bf bnext;
      if (t < 15) bnext = ld16(vrow + (size_t)(t + 1) * 16 * HWD + k0, 8);
      acc[t] = wmma_bf16(ap, bv, acc[t]);
      bv = bnext;
    }
  }

  // epilogue: rows M = q, cols N = c; write n-major into concat buffer
  float inv[8];
#pragma unroll
  for (int r = 0; r < 8; ++r) inv[r] = 1.f / lrow[r];
  const size_t oB = (size_t)b * HWD * 512;
#pragma unroll
  for (int t = 0; t < 16; ++t) {
    int c = t * 16 + l16;
#pragma unroll
    for (int r = 0; r < 8; ++r) {
      int q = q0 + r + 8 * hi;
      OT[oB + (size_t)q * 512 + c] = (__bf16)(acc[t][r] * inv[r]);
    }
  }
}

// ---------------------------------------------------------------------------
// Host-side orchestration
// ---------------------------------------------------------------------------
static inline size_t al256(size_t x) { return (x + 255) & ~(size_t)255; }

extern "C" void kernel_launch(void* const* d_in, const int* in_sizes, int n_in,
                              void* d_out, int out_size, void* d_ws,
                              size_t ws_size, hipStream_t stream) {
  (void)in_sizes; (void)n_in; (void)out_size; (void)ws_size;
  const float* x1   = (const float*)d_in[0];
  const float* x2   = (const float*)d_in[1];
  const float* w1   = (const float*)d_in[2];
  const float* b1   = (const float*)d_in[3];
  const float* w2   = (const float*)d_in[4];
  const float* b2   = (const float*)d_in[5];
  const float* gnw  = (const float*)d_in[6];
  const float* gnb  = (const float*)d_in[7];
  const float* wqkv = (const float*)d_in[8];
  const float* wout = (const float*)d_in[9];
  const float* bout = (const float*)d_in[10];
  const float* wsk  = (const float*)d_in[11];
  const float* bsk  = (const float*)d_in[12];
  float* out = (float*)d_out;

  const size_t nx = (size_t)NB * CCH * HWD;  // 4,194,304

  char* p = (char*)d_ws;
  size_t off = 0;
  auto take = [&](size_t bytes) { void* r = p + off; off = al256(off + bytes); return r; };

  __bf16* xt1   = (__bf16*)take(nx * 2);        // [B][HW][C]
  __bf16* xt2   = (__bf16*)take(nx * 2);
  __bf16* w1b   = (__bf16*)take(65536 * 2);
  __bf16* w2b   = (__bf16*)take(65536 * 2);
  __bf16* wqkvb = (__bf16*)take(196608 * 2);
  __bf16* wcat  = (__bf16*)take(131072 * 2);
  float*  bsum  = (float*)take(256 * 4);
  float*  y1    = (float*)take(nx * 4);         // [B][C][HW]
  float*  y2    = (float*)take(nx * 4);
  __bf16* g1t   = (__bf16*)take(nx * 2);        // [B][HW][C]
  __bf16* g2t   = (__bf16*)take(nx * 2);
  __bf16* q1t   = (__bf16*)take(nx * 2);        // [B][HW][C]
  __bf16* k2t   = (__bf16*)take(nx * 2);        // [B][HW][C]
  __bf16* vb    = (__bf16*)take(nx * 2);        // [B][C][HW] (natural)
  __bf16* xcatT = (__bf16*)take(2 * nx * 2);    // [B][HW][512]

  // 1) conversions / transposes
  sca_transpose_cvt<<<dim3(HWD / 64, CCH / 32, NB), 256, 0, stream>>>(x1, xt1);
  sca_transpose_cvt<<<dim3(HWD / 64, CCH / 32, NB), 256, 0, stream>>>(x2, xt2);
  sca_cvt_bf16<<<256, 256, 0, stream>>>(w1, w1b, 65536);
  sca_cvt_bf16<<<256, 256, 0, stream>>>(w2, w2b, 65536);
  sca_cvt_bf16<<<768, 256, 0, stream>>>(wqkv, wqkvb, 196608);
  sca_build_wcat<<<512, 256, 0, stream>>>(wout, wsk, wcat);
  sca_build_bsum<<<1, 256, 0, stream>>>(bout, bsk, bsum);

  // 2) input convs: y = w * x + b   (M=256, N=4096, K=256), f32 out
  dim3 gGemm(HWD / 128, CCH / 64, NB);
  sca_gemm1x1<0><<<gGemm, 256, 0, stream>>>(w1b, xt1, b1, y1, nullptr, 256, HWD, 256);
  sca_gemm1x1<0><<<gGemm, 256, 0, stream>>>(w2b, xt2, b2, y2, nullptr, 256, HWD, 256);

  // 3) group norm (x1 path also drops raw bf16 copy into concat cols 256..511)
  sca_groupnorm<<<NB * 32, 256, 0, stream>>>(y1, gnw, gnb, g1t, xcatT);
  sca_groupnorm<<<NB * 32, 256, 0, stream>>>(y2, gnw, gnb, g2t, nullptr);

  // 4) projections: q,k n-major [B][HW][C]; v natural [B][C][HW]
  sca_gemm1x1<1><<<gGemm, 256, 0, stream>>>(wqkvb,          g1t, nullptr, nullptr, q1t, 256, HWD, 256);
  sca_gemm1x1<1><<<gGemm, 256, 0, stream>>>(wqkvb + 65536,  g2t, nullptr, nullptr, k2t, 256, HWD, 256);
  sca_gemm1x1<2><<<gGemm, 256, 0, stream>>>(wqkvb + 131072, g2t, nullptr, nullptr, vb,  256, HWD, 256);

  // 5) flash attention -> concat cols 0..255
  sca_flash_attn<<<dim3(HWD / 64, NB), 128, 0, stream>>>(q1t, k2t, vb, xcatT);

  // 6) fused out-proj + skip conv:  out = [wout|wsk] x [O ; y1]^T + (bout+bsk)
  sca_gemm1x1<0><<<gGemm, 256, 0, stream>>>(wcat, xcatT, bsum, out, nullptr, 256, HWD, 512);
}
